// Matcher_62354335203809
// MI455X (gfx1250) — compile-verified, matrix operand order checked by reading, not run on hardware
//
#include <hip/hip_runtime.h>

// CDNA5 / gfx1250: wave32, WMMA 16x16x32 bf16 (f32 accumulate)
typedef __attribute__((ext_vector_type(16))) __bf16 v16bf;
typedef __attribute__((ext_vector_type(8)))  float  v8f;

#define Q_DIM   128
#define K_DIM   2048
#define T_DIM   32          // seq_len
#define Z_DIM   24          // d5 / seq_len
#define K_PER_WAVE 8
#define WAVES_PER_BLOCK 8

// Load one 16x32 bf16 WMMA fragment (A layout; B has the identical per-lane
// gather since B is column-per-lane and our B == memory rows) from a
// row-major [16 rows x 24 floats] block. K=24..31 is zero padding.
//   lanes 0..15 : row M=lane,    K = {0..7, 16..23}
//   lanes 16..31: row M=lane-16, K = {8..15, 24..31(=0)}
__device__ __forceinline__ v16bf load_frag_rows24(const float* __restrict__ rowbase,
                                                  int lane) {
  const int  m  = lane & 15;
  const bool hi = lane >= 16;
  const float* p = rowbase + m * Z_DIM;            // 96B pitch, 16B aligned
  const int c0 = hi ? 8 : 0;
  float4 f0 = *(const float4*)(p + c0);            // K 0..3   / 8..11
  float4 f1 = *(const float4*)(p + c0 + 4);        // K 4..7   / 12..15
  float4 f2 = *(const float4*)(p + 16);            // K 16..19 / (pad)
  float4 f3 = *(const float4*)(p + 20);            // K 20..23 / (pad)
  if (hi) { f2 = make_float4(0.f,0.f,0.f,0.f); f3 = make_float4(0.f,0.f,0.f,0.f); }
  v16bf a;
  a[0]=(__bf16)f0.x; a[1]=(__bf16)f0.y; a[2]=(__bf16)f0.z; a[3]=(__bf16)f0.w;
  a[4]=(__bf16)f1.x; a[5]=(__bf16)f1.y; a[6]=(__bf16)f1.z; a[7]=(__bf16)f1.w;
  a[8]=(__bf16)f2.x; a[9]=(__bf16)f2.y; a[10]=(__bf16)f2.z; a[11]=(__bf16)f2.w;
  a[12]=(__bf16)f3.x; a[13]=(__bf16)f3.y; a[14]=(__bf16)f3.z; a[15]=(__bf16)f3.w;
  return a;
}

// Per-lane max over the 8 f32 accumulator rows of a D tile.
__device__ __forceinline__ float tilemax8(const v8f& t) {
  float m = fmaxf(fmaxf(fmaxf(t[0], t[1]), fmaxf(t[2], t[3])),
                  fmaxf(fmaxf(t[4], t[5]), fmaxf(t[6], t[7])));
  return m;
}

__global__ void __launch_bounds__(WAVES_PER_BLOCK * 32)
matcher_kernel(const float* __restrict__ tgt,
               const float* __restrict__ mem,
               float* __restrict__ out) {
  const int lane = threadIdx.x & 31;
  const int wave = threadIdx.x >> 5;
  const int q    = blockIdx.y;
  const int k0   = (blockIdx.x * WAVES_PER_BLOCK + wave) * K_PER_WAVE;

  // Q fragments for this q (t = 0..15 and t = 16..31), reused for all k.
  const float* qrows = tgt + (size_t)q * T_DIM * Z_DIM;
  const v16bf a0 = load_frag_rows24(qrows, lane);
  const v16bf a1 = load_frag_rows24(qrows + 16 * Z_DIM, lane);

  for (int kk = 0; kk < K_PER_WAVE; ++kk) {
    const int k = k0 + kk;
    const float* krows = mem + (size_t)k * T_DIM * Z_DIM;
    if (kk + 1 < K_PER_WAVE)                       // global_prefetch_b8
      __builtin_prefetch(krows + T_DIM * Z_DIM, 0, 1);

    const v16bf b0 = load_frag_rows24(krows, lane);              // s = 0..15
    const v16bf b1 = load_frag_rows24(krows + 16 * Z_DIM, lane); // s = 16..31

    // S = Q x K^T tiles: M = t, N = s.  One WMMA covers the whole d=24
    // contraction (K=32, padded).
    v8f s00 = {}, s01 = {}, s10 = {}, s11 = {};
    s00 = __builtin_amdgcn_wmma_f32_16x16x32_bf16(false, a0, false, b0,
                                                  (short)0, s00, false, false);
    s01 = __builtin_amdgcn_wmma_f32_16x16x32_bf16(false, a0, false, b1,
                                                  (short)0, s01, false, false);
    s10 = __builtin_amdgcn_wmma_f32_16x16x32_bf16(false, a1, false, b0,
                                                  (short)0, s10, false, false);
    s11 = __builtin_amdgcn_wmma_f32_16x16x32_bf16(false, a1, false, b1,
                                                  (short)0, s11, false, false);

    // S^T = K x Q^T tiles (operands swapped): M = s, N = t.  These extra
    // WMMAs move the "max over s" reduction onto the per-lane register axis,
    // eliminating the per-row cross-lane butterflies (DS traffic).
    v8f t00 = {}, t01 = {}, t10 = {}, t11 = {};   // t[su][tv]
    t00 = __builtin_amdgcn_wmma_f32_16x16x32_bf16(false, b0, false, a0,
                                                  (short)0, t00, false, false);
    t01 = __builtin_amdgcn_wmma_f32_16x16x32_bf16(false, b0, false, a1,
                                                  (short)0, t01, false, false);
    t10 = __builtin_amdgcn_wmma_f32_16x16x32_bf16(false, b1, false, a0,
                                                  (short)0, t10, false, false);
    t11 = __builtin_amdgcn_wmma_f32_16x16x32_bf16(false, b1, false, a1,
                                                  (short)0, t11, false, false);

    // rowmax: max over s for t = tv*16 + (lane&15).  Per-lane max over the
    // 16 s-values in this half (su tiles x 8 regs), then half-swap max.
    float mrow0 = fmaxf(tilemax8(t00), tilemax8(t10));  // t = 0..15
    float mrow1 = fmaxf(tilemax8(t01), tilemax8(t11));  // t = 16..31
    mrow0 = fmaxf(mrow0, __shfl_xor(mrow0, 16, 32));
    mrow1 = fmaxf(mrow1, __shfl_xor(mrow1, 16, 32));

    // colmax: max over t for s = sn*16 + (lane&15).  Same pattern on S.
    float cm0 = fmaxf(tilemax8(s00), tilemax8(s10));    // s = 0..15
    float cm1 = fmaxf(tilemax8(s01), tilemax8(s11));    // s = 16..31
    cm0 = fmaxf(cm0, __shfl_xor(cm0, 16, 32));
    cm1 = fmaxf(cm1, __shfl_xor(cm1, 16, 32));

    // Single shared butterfly-add: sum of 32 row maxima + 32 col maxima.
    float sumv = (mrow0 + mrow1) + (cm0 + cm1);
#pragma unroll
    for (int off = 1; off < 16; off <<= 1)
      sumv += __shfl_xor(sumv, off, 32);

    const float mean = sumv * (1.0f / 64.0f);
    const float p = 1.0f / (1.0f + __expf(-mean));
    if (lane == 0) {
      out[(size_t)q * K_DIM + k] = p;                               // tuple[0]
      out[(size_t)Q_DIM * K_DIM + (size_t)q * K_DIM + k] = p;       // tuple[1]
    }
  }
}

extern "C" void kernel_launch(void* const* d_in, const int* in_sizes, int n_in,
                              void* d_out, int out_size, void* d_ws, size_t ws_size,
                              hipStream_t stream) {
  const float* tgt = (const float*)d_in[0];   // [128, 768] f32
  const float* mem = (const float*)d_in[1];   // [2048, 768] f32
  (void)in_sizes; (void)n_in; (void)d_ws; (void)ws_size; (void)out_size;
  float* out = (float*)d_out;                 // [2 * 128 * 2048] f32

  dim3 grid(K_DIM / (WAVES_PER_BLOCK * K_PER_WAVE), Q_DIM); // (32, 128)
  dim3 block(WAVES_PER_BLOCK * 32);                          // 256 = 8 waves
  matcher_kernel<<<grid, block, 0, stream>>>(tgt, mem, out);
}
